// MultiheadPositionalAttention_35948876268137
// MI455X (gfx1250) — matrix operand order repaired and checked by reading, not compiled, not run on hardware
//
#include <hip/hip_runtime.h>
#include <hip/hip_bf16.h>
#include <math.h>

#define NN 2048
#define BB 4
#define CC 256
#define HH 8
#define HD 32
#define KNN 5
#define NEGV (-1000000000.0f)

typedef __attribute__((ext_vector_type(16))) __bf16 v16bf;
typedef __attribute__((ext_vector_type(8)))  __bf16 bf16x8;
typedef __attribute__((ext_vector_type(8)))  float  v8f;

// ---------- fragment helpers (wave32, 16x16x32 bf16 WMMA layouts) ----------
// A (16x32): lane holds row M=lane%16; K = 16*half + e
__device__ inline v16bf a_frag_from_f32(const float* __restrict__ row, int half) {
  v16bf a;
  const float* p0 = row + half * 16;
  const float* p1 = row + half * 16 + 8;
#pragma unroll
  for (int e = 0; e < 8; ++e) { a[e] = (__bf16)p0[e]; a[e + 8] = (__bf16)p1[e]; }
  return a;
}
__device__ inline v16bf a_frag_from_bf16(const __bf16* __restrict__ row, int half) {
  bf16x8 lo = *(const bf16x8*)(row + half * 16);
  bf16x8 hi = *(const bf16x8*)(row + half * 16 + 8);
  v16bf a;
#pragma unroll
  for (int e = 0; e < 8; ++e) { a[e] = lo[e]; a[e + 8] = hi[e]; }
  return a;
}
// B (32x16): lane holds col N=lane%16; K = 16*half + e  (caller passes pointer at K=16*half)
__device__ inline v16bf b_frag_from_f32(const float* __restrict__ run16) {
  v16bf b;
#pragma unroll
  for (int e = 0; e < 16; ++e) b[e] = (__bf16)run16[e];
  return b;
}
__device__ inline v16bf b_frag_from_bf16(const __bf16* __restrict__ run16) {
  bf16x8 lo = *(const bf16x8*)(run16);
  bf16x8 hi = *(const bf16x8*)(run16 + 8);
  v16bf b;
#pragma unroll
  for (int e = 0; e < 8; ++e) { b[e] = lo[e]; b[e + 8] = hi[e]; }
  return b;
}

// ---------- 1) K-nearest targets per (b, source j) ----------
__global__ __launch_bounds__(256) void knn_kernel(const float* __restrict__ srcp,
                                                  const float* __restrict__ tgtp,
                                                  int* __restrict__ idx) {
  int t = blockIdx.x * 256 + threadIdx.x;
  if (t >= BB * NN) return;
  int b = t / NN, j = t % NN;
  float sx = srcp[(j * BB + b) * 3 + 0];
  float sy = srcp[(j * BB + b) * 3 + 1];
  float sz = srcp[(j * BB + b) * 3 + 2];
  float bd[KNN]; int bi[KNN];
#pragma unroll
  for (int q = 0; q < KNN; ++q) { bd[q] = 3.4e38f; bi[q] = -1; }
  for (int i = 0; i < NN; ++i) {
    const float* tp = tgtp + (i * BB + b) * 3;
    float dx = tp[0] - sx, dy = tp[1] - sy, dz = tp[2] - sz;
    float d = dx * dx + dy * dy + dz * dz;
    if (d < bd[KNN - 1]) {
      int p = KNN - 1;
      while (p > 0 && d < bd[p - 1]) { bd[p] = bd[p - 1]; bi[p] = bi[p - 1]; --p; }
      bd[p] = d; bi[p] = i;
    }
  }
#pragma unroll
  for (int q = 0; q < KNN; ++q) idx[t * 8 + q] = bi[q];
#pragma unroll
  for (int q = KNN; q < 8; ++q) idx[t * 8 + q] = -1;
}

// ---------- 2) QKV projection GEMM (bf16 WMMA), writes packed bf16 ----------
__global__ __launch_bounds__(32) void proj_kernel(const float* __restrict__ q,
                                                  const float* __restrict__ k,
                                                  const float* __restrict__ v,
                                                  const float* __restrict__ w,
                                                  const float* __restrict__ bias,
                                                  __bf16* __restrict__ qv,
                                                  __bf16* __restrict__ kv,
                                                  __bf16* __restrict__ vv) {
  int it = blockIdx.x, dt = blockIdx.y, o = blockIdx.z;
  int lane = threadIdx.x, half = lane >> 4, lp = lane & 15;
  const float* X = (o == 0) ? q : (o == 1) ? k : v;
  const float* W = w + o * CC * CC;
  const float* Bp = bias + o * CC;
  int i0 = it * 16, d0 = dt * 16;
  const float* xrow = X + (i0 + lp) * CC;
  const float* wrow = W + (d0 + lp) * CC;
  v8f acc = {};
#pragma unroll
  for (int k0 = 0; k0 < CC; k0 += 32) {
    v16bf a  = a_frag_from_f32(xrow + k0, half);
    v16bf bf = b_frag_from_f32(wrow + k0 + half * 16);
    acc = __builtin_amdgcn_wmma_f32_16x16x32_bf16(false, a, false, bf, (short)0, acc, false, false);
  }
  const float scale = 0.17677669529663687f; // 1/sqrt(32)
#pragma unroll
  for (int r = 0; r < 8; ++r) {
    int row = i0 + r + half * 8;        // flattened (n*B + b)
    int col = d0 + lp;                  // channel
    int n = row / BB, bb = row % BB;
    int h = col / HD, dd = col % HD;
    float val = acc[r] + Bp[col];
    if (o == 0)      qv[(((size_t)(bb * HH) + h) * NN + n) * HD + dd] = (__bf16)(val * scale);
    else if (o == 1) kv[(((size_t)(bb * HH) + h) * NN + n) * HD + dd] = (__bf16)val;
    else             vv[(((size_t)(bb * HH) + h) * HD + dd) * NN + n] = (__bf16)val;
  }
}

// ---------- score tile helper: S = q_frag x k_frag, masked ----------
__device__ inline void score_tile(const v16bf& aq, const __bf16* __restrict__ kbase,
                                  const int* __restrict__ idx, const unsigned char* __restrict__ kpm,
                                  int b, int i0, int j0, int jt, int half, int lp,
                                  float* __restrict__ wt) {
  int j = j0 + jt * 16 + lp;
  v16bf bk = b_frag_from_bf16(kbase + (size_t)j * HD + half * 16);
  v8f s = {};
  s = __builtin_amdgcn_wmma_f32_16x16x32_bf16(false, aq, false, bk, (short)0, s, false, false);
  const int* ij = idx + ((size_t)b * NN + j) * 8;
  int x0 = ij[0], x1 = ij[1], x2 = ij[2], x3 = ij[3], x4 = ij[4];
  bool pad = kpm[b * NN + j] != 0;
#pragma unroll
  for (int r = 0; r < 8; ++r) {
    int i = i0 + r + half * 8;
    float sv = s[r];
    bool sel = (i == x0) | (i == x1) | (i == x2) | (i == x3) | (i == x4);
    sv += sel ? 0.0f : NEGV;
    if (pad) sv = -INFINITY;
    wt[(r + half * 8) * 32 + jt * 16 + lp] = sv;
  }
}

// prefetch next j-chunk of K rows + idx into WGP$ (global_prefetch_b8)
__device__ inline void prefetch_next_chunk(const __bf16* __restrict__ kbase,
                                           const int* __restrict__ idx,
                                           int b, int j0n, int lane) {
  if (j0n < NN) {
    __builtin_prefetch(kbase + (size_t)(j0n + lane) * HD, 0, 3);
    __builtin_prefetch(idx + ((size_t)b * NN + j0n + lane) * 8, 0, 3);
  }
}

// ---------- 3) pass 1: row softmax stats (online max / denom) ----------
__global__ __launch_bounds__(256) void stats_kernel(const __bf16* __restrict__ qv,
                                                    const __bf16* __restrict__ kv,
                                                    const int* __restrict__ idx,
                                                    const unsigned char* __restrict__ kpm,
                                                    float* __restrict__ mrow,
                                                    float* __restrict__ lrow) {
  __shared__ float tile[8 * 512];
  int b  = blockIdx.x / (NN / 16);
  int i0 = (blockIdx.x % (NN / 16)) * 16;
  int w = threadIdx.x >> 5, lane = threadIdx.x & 31;
  int half = lane >> 4, lp = lane & 15;
  int h = w;
  const __bf16* qbase = qv + (((size_t)(b * HH) + h) * NN) * HD;
  const __bf16* kbase = kv + (((size_t)(b * HH) + h) * NN) * HD;
  v16bf aq = a_frag_from_bf16(qbase + (size_t)(i0 + lp) * HD, half);
  float m_run = -INFINITY, l_run = 0.0f;
  float* wt = tile + w * 512;
  for (int j0 = 0; j0 < NN; j0 += 32) {
    prefetch_next_chunk(kbase, idx, b, j0 + 32, lane);
    score_tile(aq, kbase, idx, kpm, b, i0, j0, 0, half, lp, wt);
    score_tile(aq, kbase, idx, kpm, b, i0, j0, 1, half, lp, wt);
    __syncthreads();
    if (lane < 16) {
      const float* rowp = wt + lane * 32;
      float cm = -INFINITY;
#pragma unroll
      for (int c = 0; c < 32; ++c) cm = fmaxf(cm, rowp[c]);
      float mn = fmaxf(m_run, cm);
      if (mn != -INFINITY) {
        float accs = 0.0f;
#pragma unroll
        for (int c = 0; c < 32; ++c) accs += __expf(rowp[c] - mn);
        l_run = l_run * __expf(m_run - mn) + accs;
        m_run = mn;
      }
    }
    __syncthreads();
  }
  if (lane < 16) {
    mrow[((size_t)(b * HH) + h) * NN + i0 + lane] = m_run;
    lrow[((size_t)(b * HH) + h) * NN + i0 + lane] = l_run;
  }
}

// ---------- 4) pass 2: probs, head-averaged attn output, PV GEMM ----------
__global__ __launch_bounds__(256) void attn_kernel(const __bf16* __restrict__ qv,
                                                   const __bf16* __restrict__ kv,
                                                   const __bf16* __restrict__ vv,
                                                   const int* __restrict__ idx,
                                                   const unsigned char* __restrict__ kpm,
                                                   const float* __restrict__ mrow,
                                                   const float* __restrict__ lrow,
                                                   float* __restrict__ attn_out,
                                                   __bf16* __restrict__ ctx) {
  __shared__ float tile[8 * 512];
  __shared__ float sm[8 * 16];
  __shared__ float sil[8 * 16];
  int b  = blockIdx.x / (NN / 16);
  int i0 = (blockIdx.x % (NN / 16)) * 16;
  int w = threadIdx.x >> 5, lane = threadIdx.x & 31;
  int half = lane >> 4, lp = lane & 15;
  int h = w;
  if (lane < 16) {
    float mv = mrow[((size_t)(b * HH) + h) * NN + i0 + lane];
    float lv = lrow[((size_t)(b * HH) + h) * NN + i0 + lane];
    sm[w * 16 + lane]  = mv;
    sil[w * 16 + lane] = 1.0f / lv;
  }
  __syncthreads();
  const __bf16* qbase = qv + (((size_t)(b * HH) + h) * NN) * HD;
  const __bf16* kbase = kv + (((size_t)(b * HH) + h) * NN) * HD;
  const __bf16* vbase = vv + (((size_t)(b * HH) + h) * HD) * NN;
  v16bf aq = a_frag_from_bf16(qbase + (size_t)(i0 + lp) * HD, half);
  v8f acc0 = {}, acc1 = {};
  float* wt = tile + w * 512;
  for (int j0 = 0; j0 < NN; j0 += 32) {
    // prefetch next chunk's K rows, idx and V row-segments into WGP$
    prefetch_next_chunk(kbase, idx, b, j0 + 32, lane);
    if (j0 + 32 < NN) {
      __builtin_prefetch(vbase + (size_t)lane * NN + j0 + 32, 0, 3);
    }
    score_tile(aq, kbase, idx, kpm, b, i0, j0, 0, half, lp, wt);
    score_tile(aq, kbase, idx, kpm, b, i0, j0, 1, half, lp, wt);
    // convert in place to normalized probabilities
#pragma unroll
    for (int jt = 0; jt < 2; ++jt) {
#pragma unroll
      for (int r = 0; r < 8; ++r) {
        int M = r + half * 8;
        float sv = wt[M * 32 + jt * 16 + lp];
        wt[M * 32 + jt * 16 + lp] = __expf(sv - sm[w * 16 + M]) * sil[w * 16 + M];
      }
    }
    __syncthreads();
    // deterministic 8-head mean -> attn output
    {
      int t = threadIdx.x;
#pragma unroll
      for (int rep = 0; rep < 2; ++rep) {
        int e = t + rep * 256;
        float ssum = 0.0f;
#pragma unroll
        for (int ww = 0; ww < 8; ++ww) ssum += tile[ww * 512 + e];
        int rr = e >> 5, cc = e & 31;
        attn_out[(((size_t)b * NN) + (i0 + rr)) * NN + j0 + cc] = ssum * 0.125f;
      }
    }
    // rebuild prob tile as bf16 A-fragment, multiply by V (two 16-wide d tiles)
    v16bf ap;
    {
      const float* rowp = wt + lp * 32;
#pragma unroll
      for (int e = 0; e < 8; ++e) {
        ap[e]     = (__bf16)rowp[half * 16 + e];
        ap[e + 8] = (__bf16)rowp[half * 16 + 8 + e];
      }
    }
    v16bf bv0 = b_frag_from_bf16(vbase + (size_t)(0 + lp) * NN + j0 + half * 16);
    v16bf bv1 = b_frag_from_bf16(vbase + (size_t)(16 + lp) * NN + j0 + half * 16);
    acc0 = __builtin_amdgcn_wmma_f32_16x16x32_bf16(false, ap, false, bv0, (short)0, acc0, false, false);
    acc1 = __builtin_amdgcn_wmma_f32_16x16x32_bf16(false, ap, false, bv1, (short)0, acc1, false, false);
    __syncthreads();
  }
#pragma unroll
  for (int r = 0; r < 8; ++r) {
    int i = i0 + r + half * 8;
    ctx[((size_t)i * BB + b) * CC + h * HD + lp]      = (__bf16)acc0[r];
    ctx[((size_t)i * BB + b) * CC + h * HD + 16 + lp] = (__bf16)acc1[r];
  }
}

// ---------- 5) output projection GEMM ----------
__global__ __launch_bounds__(32) void outproj_kernel(const __bf16* __restrict__ ctx,
                                                     const float* __restrict__ ow,
                                                     const float* __restrict__ ob,
                                                     float* __restrict__ out) {
  int it = blockIdx.x, dt = blockIdx.y;
  int lane = threadIdx.x, half = lane >> 4, lp = lane & 15;
  int i0 = it * 16, d0 = dt * 16;
  const __bf16* arow = ctx + (size_t)(i0 + lp) * CC;
  const float*  wrow = ow + (d0 + lp) * CC;
  v8f acc = {};
#pragma unroll
  for (int k0 = 0; k0 < CC; k0 += 32) {
    v16bf a  = a_frag_from_bf16(arow + k0, half);
    v16bf bf = b_frag_from_f32(wrow + k0 + half * 16);
    acc = __builtin_amdgcn_wmma_f32_16x16x32_bf16(false, a, false, bf, (short)0, acc, false, false);
  }
#pragma unroll
  for (int r = 0; r < 8; ++r) {
    int row = i0 + r + half * 8;
    int col = d0 + lp;
    out[(size_t)row * CC + col] = acc[r] + ob[col];
  }
}

extern "C" void kernel_launch(void* const* d_in, const int* in_sizes, int n_in,
                              void* d_out, int out_size, void* d_ws, size_t ws_size,
                              hipStream_t stream) {
  const float* query = (const float*)d_in[0];
  const float* key   = (const float*)d_in[1];
  const float* value = (const float*)d_in[2];
  const float* srcp  = (const float*)d_in[3];
  const float* tgtp  = (const float*)d_in[4];
  const unsigned char* kpm = (const unsigned char*)d_in[5];
  const float* in_w  = (const float*)d_in[6];
  const float* in_b  = (const float*)d_in[7];
  const float* out_w = (const float*)d_in[8];
  const float* out_b = (const float*)d_in[9];

  char* ws = (char*)d_ws;
  __bf16* qv  = (__bf16*)(ws);                       // 4 MiB: [B][H][N][32] bf16 (pre-scaled)
  __bf16* kv  = (__bf16*)(ws + (4u << 20));          // 4 MiB: [B][H][N][32]
  __bf16* vv  = (__bf16*)(ws + (8u << 20));          // 4 MiB: [B][H][32][N] (transposed)
  __bf16* ctx = (__bf16*)(ws + (12u << 20));         // 4 MiB: [N*B][256]
  int*    idx = (int*)  (ws + (16u << 20));          // 256 KiB: [B][N][8]
  float*  mrow = (float*)(ws + (16u << 20) + (1u << 18)); // 256 KiB: [B][H][N]
  float*  lrow = (float*)(ws + (16u << 20) + (2u << 18)); // 256 KiB

  float* out_src2 = (float*)d_out;                       // (N,B,C)
  float* out_attn = out_src2 + (size_t)NN * BB * CC;     // (B,N,N)

  knn_kernel<<<(BB * NN + 255) / 256, 256, 0, stream>>>(srcp, tgtp, idx);
  proj_kernel<<<dim3((NN * BB) / 16, CC / 16, 3), 32, 0, stream>>>(query, key, value,
                                                                   in_w, in_b, qv, kv, vv);
  stats_kernel<<<BB * (NN / 16), 256, 0, stream>>>(qv, kv, idx, kpm, mrow, lrow);
  attn_kernel<<<BB * (NN / 16), 256, 0, stream>>>(qv, kv, vv, idx, kpm, mrow, lrow,
                                                  out_attn, ctx);
  outproj_kernel<<<dim3((NN * BB) / 16, CC / 16), 32, 0, stream>>>(ctx, out_w, out_b, out_src2);
}
